// NearestEmbed_11029476016539
// MI455X (gfx1250) — matrix operand order repaired and checked by reading, not compiled, not run on hardware
//
#include <hip/hip_runtime.h>
#include <hip/hip_bf16.h>

typedef __attribute__((ext_vector_type(16))) __bf16 v16bf;
typedef __attribute__((ext_vector_type(8)))  float  v8f;

#define D_DIM   64
#define K_CODES 512
#define HW      1024
#define NTILES  (K_CODES / 16)   // 32 N-tiles of 16 codes

// ---------------------------------------------------------------------------
// Prep: emb (D x K, f32) -> embT bf16 [k][d] (k-major, contiguous 128B rows)
//       e2[k] = sum_d emb[d][k]^2  (f32)
// ---------------------------------------------------------------------------
__global__ __launch_bounds__(256) void vq_prep(const float* __restrict__ emb,
                                               __bf16* __restrict__ embT,
                                               float* __restrict__ e2) {
  int k = blockIdx.x * blockDim.x + threadIdx.x;
  if (k >= K_CODES) return;
  float s = 0.f;
#pragma unroll
  for (int d = 0; d < D_DIM; ++d) {
    float v = emb[d * K_CODES + k];
    s += v * v;
    embT[k * D_DIM + d] = (__bf16)v;
  }
  e2[k] = s;
}

// ---------------------------------------------------------------------------
// Main: per 16-position tile, scores = e2[k] - 2 * x . e_k over all 512 codes
// via v_wmma_f32_16x16x32_bf16 (two independent accumulator chains per
// iteration for XDL overlap), fused row-argmin + f32 codebook gather.
// 4 waves / block, one M-tile per wave.
// ---------------------------------------------------------------------------
__global__ __launch_bounds__(128) void vq_main(const float* __restrict__ x,
                                               const float* __restrict__ emb,
                                               const __bf16* __restrict__ embT,
                                               const float* __restrict__ e2,
                                               float* __restrict__ out,
                                               float* __restrict__ outIdx) {
  __shared__ float s_sc[4][8][32];
  __shared__ int   s_ki[4][8][32];
  __shared__ float s_ps[4][32];
  __shared__ int   s_pk[4][32];
  __shared__ int   s_fk[4][16];

  const int lane = threadIdx.x & 31;
  const int wave = threadIdx.x >> 5;
  const int g    = lane >> 4;    // half-wave (A/B layout group)
  const int hl   = lane & 15;    // lane within half

  const int tile = blockIdx.x * 4 + wave;   // 4096 tiles total
  const int p0   = tile * 16;               // first position of this tile
  const int b    = p0 >> 10;                // batch (1024 positions per batch)
  const int hw0  = p0 & 1023;               // spatial offset within batch
  const float* xb = x + (size_t)b * (D_DIM * HW);

  // --- A fragments (16x32 bf16 each, K = d-dimension) ---
  // A layout (contiguous, matches B): lane half g, element e -> K = 16*g + e
  v16bf a0, a1;
#pragma unroll
  for (int e = 0; e < 16; ++e) {
    int d0 = e + 16 * g;
    a0[e] = (__bf16)xb[(size_t)d0 * HW + hw0 + hl];          // K = 0..31
    a1[e] = (__bf16)xb[(size_t)(d0 + 32) * HW + hw0 + hl];   // K = 32..63
  }

  // Per-lane running argmin over this lane's column set {t*16 + hl}
  float best[8];
  int   bestk[8];
#pragma unroll
  for (int r = 0; r < 8; ++r) { best[r] = 3.4e38f; bestk[r] = 0; }

  for (int t = 0; t < NTILES; t += 2) {
    const int k0 = t * 16 + hl;
    const int k1 = k0 + 16;
    // B layout: lanes 0-15 hold K=0..15, lanes 16-31 hold K=16..31, N = hl.
    // embT row k is 64 contiguous bf16 -> per-lane 32B contiguous loads.
    const __bf16* bb0 = embT + (size_t)k0 * D_DIM + 16 * g;
    const __bf16* bb1 = embT + (size_t)k1 * D_DIM + 16 * g;
    v16bf b00 = *(const v16bf*)(bb0);        // tile t,   d = 16g + e
    v16bf b01 = *(const v16bf*)(bb0 + 32);   // tile t,   d = 32 + 16g + e
    v16bf b10 = *(const v16bf*)(bb1);        // tile t+1
    v16bf b11 = *(const v16bf*)(bb1 + 32);
    const float e20 = e2[k0];
    const float e21 = e2[k1];

    v8f acc0 = {0.f, 0.f, 0.f, 0.f, 0.f, 0.f, 0.f, 0.f};
    v8f acc1 = {0.f, 0.f, 0.f, 0.f, 0.f, 0.f, 0.f, 0.f};
    // Two independent WMMA chains -> back-to-back XDL issue.
    acc0 = __builtin_amdgcn_wmma_f32_16x16x32_bf16(false, a0, false, b00,
                                                   (short)0, acc0, false, false);
    acc1 = __builtin_amdgcn_wmma_f32_16x16x32_bf16(false, a0, false, b10,
                                                   (short)0, acc1, false, false);
    acc0 = __builtin_amdgcn_wmma_f32_16x16x32_bf16(false, a1, false, b01,
                                                   (short)0, acc0, false, false);
    acc1 = __builtin_amdgcn_wmma_f32_16x16x32_bf16(false, a1, false, b11,
                                                   (short)0, acc1, false, false);

#pragma unroll
    for (int r = 0; r < 8; ++r) {          // row m = g*8 + r, col n = hl
      float s0 = fmaf(-2.f, acc0[r], e20);
      if (s0 < best[r]) { best[r] = s0; bestk[r] = k0; }  // strict < keeps smallest k
    }
#pragma unroll
    for (int r = 0; r < 8; ++r) {
      float s1 = fmaf(-2.f, acc1[r], e21);
      if (s1 < best[r]) { best[r] = s1; bestk[r] = k1; }
    }
  }

  // --- Cross-lane row reduction via LDS (tie-break: smaller k wins) ---
#pragma unroll
  for (int r = 0; r < 8; ++r) {
    s_sc[wave][r][lane] = best[r];
    s_ki[wave][r][lane] = bestk[r];
  }
  __syncthreads();

  {
    const int m  = lane & 15;   // row this lane reduces
    const int jj = lane >> 4;   // which 8 of the row's 16 candidates
    float bs = 3.4e38f; int bk = 0x7fffffff;
#pragma unroll
    for (int i = 0; i < 8; ++i) {
      int src = ((m >> 3) << 4) + (jj << 3) + i;  // lanes holding row m
      float s = s_sc[wave][m & 7][src];
      int   k = s_ki[wave][m & 7][src];
      if (s < bs || (s == bs && k < bk)) { bs = s; bk = k; }
    }
    s_ps[wave][lane] = bs;
    s_pk[wave][lane] = bk;
  }
  __syncthreads();

  if (lane < 16) {
    float s0v = s_ps[wave][lane];      int k0 = s_pk[wave][lane];
    float s1v = s_ps[wave][lane + 16]; int k1 = s_pk[wave][lane + 16];
    int kf = (s1v < s0v || (s1v == s0v && k1 < k0)) ? k1 : k0;
    s_fk[wave][lane] = kf;
    outIdx[p0 + lane] = (float)kf;     // argmin, stored in f32 output buffer
  }
  __syncthreads();

  // --- Gather winning f32 code columns into channel-major output ---
  {
    const int m   = lane & 15;
    const int kb  = s_fk[wave][m];
    const int hw  = hw0 + m;
    float* outb = out + (size_t)b * (D_DIM * HW) + hw;
    const int dbase = (lane >> 4) * 32;   // two lanes per row split d-range
#pragma unroll 4
    for (int dd = 0; dd < 32; ++dd) {
      int d = dbase + dd;
      outb[(size_t)d * HW] = emb[d * K_CODES + kb];  // codebook is L2-hot
    }
  }
}

// ---------------------------------------------------------------------------
extern "C" void kernel_launch(void* const* d_in, const int* in_sizes, int n_in,
                              void* d_out, int out_size, void* d_ws, size_t ws_size,
                              hipStream_t stream) {
  const float* x   = (const float*)d_in[0];   // (64, 64, 32, 32) f32
  const float* emb = (const float*)d_in[1];   // (64, 512) f32

  __bf16* embT = (__bf16*)d_ws;                         // 64 KB
  float*  e2   = (float*)((char*)d_ws + 64 * 1024);     // 2 KB

  float* out    = (float*)d_out;              // 4,194,304 f32 quantized output
  float* outIdx = out + (size_t)64 * 64 * 32 * 32;      // 65,536 argmin values

  vq_prep<<<2, 256, 0, stream>>>(emb, embT, e2);
  vq_main<<<1024, 128, 0, stream>>>(x, emb, embT, e2, out, outIdx);
}
